// LAVAMemory_52286931862194
// MI455X (gfx1250) — matrix operand order repaired and checked by reading, not compiled, not run on hardware
//
#include <hip/hip_runtime.h>
#include <hip/hip_bf16.h>
#include <math.h>

// ---------------------------------------------------------------------------
// LAVA memory layer for MI455X (gfx1250, wave32, WMMA).
// Two 16384x2048x2048 GEMMs in bf16 WMMA with f32 accumulation, fed by
// double-buffered async global->LDS tile staging (ASYNCcnt path), plus a
// fused score/top-4/softmax/gather kernel.
//
// Workspace layout (requires ~209 MB of d_ws):
//   q_bf16    : 16384*2048 bf16  (67,108,864 B)  offset 0
//   mem_bf16  : 16384*2048 bf16  (67,108,864 B)  offset 67108864
//   x_bf16    : 16384*2048 bf16  (67,108,864 B)  offset 134217728
//   Waddr_bf16:  2048*2048 bf16  ( 8,388,608 B)  offset 201326592
//   Wread_bf16:  2048*2048 bf16  ( 8,388,608 B)  offset 209715200
//   knorm_bf16:   128*2048 bf16  (   524,288 B)  offset 218103808
//   qrnorm    : 16384 f32        (    65,536 B)  offset 218628096
// ---------------------------------------------------------------------------

#define H_DIM 2048
#define N_MEM 128
#define T_TOK 16384   // B*S = 4*4096
#define LDS_STRIDE 40 // halves per LDS tile row (32 data + 8 pad, 20-dword stride)

typedef __attribute__((ext_vector_type(16))) __bf16 v16bf;
typedef __attribute__((ext_vector_type(8)))  float  v8f;
typedef int v4i_raw __attribute__((vector_size(16)));   // true vector type for builtins

union BF16x16 { uint4 u[2]; unsigned short s[16]; v16bf v; };
union BF16x8  { uint4 u;    unsigned short s[8]; };

static __device__ __forceinline__ unsigned short f2b(float f) {
    unsigned int u = __float_as_uint(f);
    u += 0x7FFFu + ((u >> 16) & 1u);          // round-to-nearest-even
    return (unsigned short)(u >> 16);
}
static __device__ __forceinline__ float b2f_lo(unsigned int w) {
    return __uint_as_float(w << 16);
}
static __device__ __forceinline__ float b2f_hi(unsigned int w) {
    return __uint_as_float(w & 0xFFFF0000u);
}

// ------------------- async global->LDS copy (CDNA5 path) --------------------
#if defined(__has_builtin)
#if __has_builtin(__builtin_amdgcn_global_load_async_to_lds_b128)
#define HAVE_ASYNC_LDS 1
#endif
#if __has_builtin(__builtin_amdgcn_s_wait_asynccnt)
#define HAVE_ASYNC_WAIT_BUILTIN 1
#endif
#endif

#define AS1 __attribute__((address_space(1)))
#define AS3 __attribute__((address_space(3)))

static __device__ __forceinline__ void copy16_g2l(void* lds, const void* g) {
#if defined(HAVE_ASYNC_LDS)
    // prototype (from hipcc diagnostic): (AS1 v4i32*, AS3 v4i32*, imm offset, imm cpol)
    void* gm = const_cast<void*>(g);
    __builtin_amdgcn_global_load_async_to_lds_b128(
        (AS1 v4i_raw*)gm, (AS3 v4i_raw*)lds, /*offset=*/0, /*cpol=*/0);
#else
    *(uint4*)lds = *(const uint4*)g;          // synchronous fallback
#endif
}

template <int N>
static __device__ __forceinline__ void wait_async_le() {
#if defined(HAVE_ASYNC_LDS)
#if defined(HAVE_ASYNC_WAIT_BUILTIN)
    __builtin_amdgcn_s_wait_asynccnt(N);
#else
    asm volatile("s_wait_asynccnt %0" ::"i"(N) : "memory");
#endif
#endif
}

// A fragment (16x32 bf16): lane (lo16,hi) holds row m=lo16,
// halves i=0..15 -> k = hi*16 + i.
static __device__ __forceinline__ v16bf frag_a(const unsigned short* rowPtr, int hi) {
    BF16x16 t;
    t.u[0] = *(const uint4*)(rowPtr + hi * 16);
    t.u[1] = *(const uint4*)(rowPtr + hi * 16 + 8);
    return t.v;
}
// B fragment (32x16 bf16): lane (lo16,hi) holds col n=lo16, halves i=0..15
// -> k = hi*16 + i (32 contiguous bytes of the weight row).
static __device__ __forceinline__ v16bf frag_b(const unsigned short* p) {
    BF16x16 t;
    t.u[0] = *(const uint4*)(p);
    t.u[1] = *(const uint4*)(p + 8);
    return t.v;
}

// --------------------------- elementwise f32 -> bf16 ------------------------
__global__ __launch_bounds__(256) void cvt_f32_bf16_kernel(
    const float* __restrict__ in, unsigned short* __restrict__ out) {
    int i = (blockIdx.x * 256 + threadIdx.x) * 8;
    float4 a = *(const float4*)(in + i);
    float4 b = *(const float4*)(in + i + 4);
    BF16x8 pk;
    pk.s[0] = f2b(a.x); pk.s[1] = f2b(a.y); pk.s[2] = f2b(a.z); pk.s[3] = f2b(a.w);
    pk.s[4] = f2b(b.x); pk.s[5] = f2b(b.y); pk.s[6] = f2b(b.z); pk.s[7] = f2b(b.w);
    *(uint4*)(out + i) = pk.u;
}

// --------------------- normalize addresses -> bf16 k_norm -------------------
__global__ __launch_bounds__(256) void addr_norm_kernel(
    const float* __restrict__ addr, unsigned short* __restrict__ knorm) {
    __shared__ float red[256];
    const float* row = addr + (size_t)blockIdx.x * H_DIM;
    float s = 0.f;
    for (int i = threadIdx.x; i < H_DIM; i += 256) { float v = row[i]; s += v * v; }
    red[threadIdx.x] = s;
    __syncthreads();
    for (int off = 128; off > 0; off >>= 1) {
        if (threadIdx.x < off) red[threadIdx.x] += red[threadIdx.x + off];
        __syncthreads();
    }
    float r = 1.0f / fmaxf(sqrtf(red[0]), 1e-12f);
    unsigned short* orow = knorm + (size_t)blockIdx.x * H_DIM;
    for (int i = threadIdx.x; i < H_DIM; i += 256) orow[i] = f2b(row[i] * r);
}

// ----------------------- per-row reciprocal norm of q -----------------------
__global__ __launch_bounds__(256) void q_rnorm_kernel(
    const unsigned short* __restrict__ qb, float* __restrict__ qrn) {
    int wave = threadIdx.x >> 5, lane = threadIdx.x & 31;
    int row  = blockIdx.x * 8 + wave;
    const unsigned short* rp = qb + (size_t)row * H_DIM;
    float s = 0.f;
    for (int i = lane * 8; i < H_DIM; i += 256) {
        uint4 u = *(const uint4*)(rp + i);
        float a0 = b2f_lo(u.x), a1 = b2f_hi(u.x);
        float a2 = b2f_lo(u.y), a3 = b2f_hi(u.y);
        float a4 = b2f_lo(u.z), a5 = b2f_hi(u.z);
        float a6 = b2f_lo(u.w), a7 = b2f_hi(u.w);
        s += a0*a0 + a1*a1 + a2*a2 + a3*a3 + a4*a4 + a5*a5 + a6*a6 + a7*a7;
    }
    #pragma unroll
    for (int m = 16; m >= 1; m >>= 1) s += __shfl_xor(s, m, 32);
    if (lane == 0) qrn[row] = 1.0f / fmaxf(sqrtf(s), 1e-12f);
}

// ------------------------------ WMMA GEMM -----------------------------------
// C[M, 2048] = A[M, 2048] * W^T, A and W bf16 row-major.
// WG = 256 threads = 8 waves; wave tile 32x64; WG tile 128x128; k-step 32.
// A/B 128x32 tiles double-buffered in LDS via async global->LDS copies.
template <bool OUTBF16>
__global__ __launch_bounds__(256) void gemm_wmma_kernel(
    const unsigned short* __restrict__ Ab, const unsigned short* __restrict__ Bw,
    void* __restrict__ Cptr) {
    __shared__ unsigned short Abuf[2][128 * LDS_STRIDE];  // 2 x 10 KB
    __shared__ unsigned short Bbuf[2][128 * LDS_STRIDE];  // 2 x 10 KB

    const int tid  = threadIdx.x;
    const int lane = tid & 31;
    const int wave = tid >> 5;
    const int lo = lane & 15, hi = lane >> 4;
    const int waveM = wave & 3, waveN = wave >> 2;
    const int mBlock = blockIdx.y * 128;
    const int nBlock = blockIdx.x * 128;

    const unsigned short* Abase = Ab + (size_t)mBlock * H_DIM;
    const unsigned short* Bbase = Bw + (size_t)nBlock * H_DIM;

    // this thread's two 16B chunks of each 128x32 tile (512 chunks / 256 thr)
    const int c0row = tid >> 2,           c0k = (tid & 3) * 8;
    const int c1row = (tid + 256) >> 2,   c1k = ((tid + 256) & 3) * 8;

    v8f acc[2][4];
    const v8f vzero = {0.f, 0.f, 0.f, 0.f, 0.f, 0.f, 0.f, 0.f};
    #pragma unroll
    for (int mt = 0; mt < 2; ++mt)
        #pragma unroll
        for (int nt = 0; nt < 4; ++nt) acc[mt][nt] = vzero;

    const int NK = H_DIM / 32;

    // prologue: stage k-step 0 into buffer 0
    copy16_g2l(&Abuf[0][c0row * LDS_STRIDE + c0k], Abase + (size_t)c0row * H_DIM + c0k);
    copy16_g2l(&Bbuf[0][c0row * LDS_STRIDE + c0k], Bbase + (size_t)c0row * H_DIM + c0k);
    copy16_g2l(&Abuf[0][c1row * LDS_STRIDE + c1k], Abase + (size_t)c1row * H_DIM + c1k);
    copy16_g2l(&Bbuf[0][c1row * LDS_STRIDE + c1k], Bbase + (size_t)c1row * H_DIM + c1k);

    for (int ki = 0; ki < NK; ++ki) {
        const int cur = ki & 1;
        if (ki + 1 < NK) {                 // prefetch next k-step
            const int nxt = cur ^ 1;
            const int kb = (ki + 1) * 32;
            copy16_g2l(&Abuf[nxt][c0row * LDS_STRIDE + c0k],
                       Abase + (size_t)c0row * H_DIM + kb + c0k);
            copy16_g2l(&Bbuf[nxt][c0row * LDS_STRIDE + c0k],
                       Bbase + (size_t)c0row * H_DIM + kb + c0k);
            copy16_g2l(&Abuf[nxt][c1row * LDS_STRIDE + c1k],
                       Abase + (size_t)c1row * H_DIM + kb + c1k);
            copy16_g2l(&Bbuf[nxt][c1row * LDS_STRIDE + c1k],
                       Bbase + (size_t)c1row * H_DIM + kb + c1k);
            wait_async_le<4>();            // first 4 (current tile) complete
        } else {
            wait_async_le<0>();
        }
        __syncthreads();                   // current tile visible to all waves

        v16bf a[2];
        #pragma unroll
        for (int mt = 0; mt < 2; ++mt)
            a[mt] = frag_a(&Abuf[cur][(waveM * 32 + mt * 16 + lo) * LDS_STRIDE], hi);
        v16bf b[4];
        #pragma unroll
        for (int nt = 0; nt < 4; ++nt)
            b[nt] = frag_b(&Bbuf[cur][(waveN * 64 + nt * 16 + lo) * LDS_STRIDE + hi * 16]);
        #pragma unroll
        for (int mt = 0; mt < 2; ++mt)
            #pragma unroll
            for (int nt = 0; nt < 4; ++nt)
                acc[mt][nt] = __builtin_amdgcn_wmma_f32_16x16x32_bf16(
                    false, a[mt], false, b[nt], (short)0, acc[mt][nt],
                    false, false);
        __syncthreads();                   // done reading before next overwrite
    }

    #pragma unroll
    for (int mt = 0; mt < 2; ++mt)
        #pragma unroll
        for (int nt = 0; nt < 4; ++nt)
            #pragma unroll
            for (int r = 0; r < 8; ++r) {
                const size_t row = (size_t)(mBlock + waveM * 32 + mt * 16 + hi * 8 + r);
                const size_t col = (size_t)(nBlock + waveN * 64 + nt * 16 + lo);
                float v = acc[mt][nt][r];
                if (OUTBF16)
                    ((unsigned short*)Cptr)[row * H_DIM + col] = f2b(v);
                else
                    ((float*)Cptr)[row * H_DIM + col] = v;
            }
}

// ------------- fused scores (WMMA) + top-4 + softmax + gather ---------------
// WG = 256 threads = 8 waves; each wave owns 16 tokens; grid = T/128.
__global__ __launch_bounds__(256) void attend_kernel(
    const unsigned short* __restrict__ qb,     // [T, H] bf16
    const unsigned short* __restrict__ knorm,  // [N, H] bf16
    const float* __restrict__ qrn,             // [T]
    const float* __restrict__ contents,        // [N, H] f32
    unsigned short* __restrict__ memb) {       // [T, H] bf16 out
    __shared__ float S[8][16][N_MEM];          // 64 KB
    __shared__ int   selIdx[8][16][4];
    __shared__ float selW[8][16][4];

    const int lane = threadIdx.x & 31;
    const int wave = threadIdx.x >> 5;
    const int lo = lane & 15, hi = lane >> 4;
    const int tBase = blockIdx.x * 128 + wave * 16;

    v8f acc[8];
    const v8f vzero = {0.f, 0.f, 0.f, 0.f, 0.f, 0.f, 0.f, 0.f};
    #pragma unroll
    for (int j = 0; j < 8; ++j) acc[j] = vzero;

    const unsigned short* qrow = qb + (size_t)(tBase + lo) * H_DIM;
    for (int kb = 0; kb < H_DIM; kb += 32) {
        v16bf a = frag_a(qrow + kb, hi);
        #pragma unroll
        for (int j = 0; j < 8; ++j) {
            v16bf b = frag_b(knorm + (size_t)(j * 16 + lo) * H_DIM + kb + hi * 16);
            acc[j] = __builtin_amdgcn_wmma_f32_16x16x32_bf16(
                false, a, false, b, (short)0, acc[j], false, false);
        }
    }
    #pragma unroll
    for (int j = 0; j < 8; ++j)
        #pragma unroll
        for (int r = 0; r < 8; ++r)
            S[wave][r + hi * 8][j * 16 + lo] = acc[j][r];
    __syncthreads();

    if (lane < 16) {                      // lanes 0..15 own one token each
        const float* Srow = S[wave][lane];
        float v0 = -1e30f, v1 = -1e30f, v2 = -1e30f, v3 = -1e30f;
        int   i0 = 0, i1 = 0, i2 = 0, i3 = 0;
        for (int n = 0; n < N_MEM; ++n) {
            float s = Srow[n];
            if (s > v3) {
                if (s > v0)      { v3=v2;i3=i2; v2=v1;i2=i1; v1=v0;i1=i0; v0=s;i0=n; }
                else if (s > v1) { v3=v2;i3=i2; v2=v1;i2=i1; v1=s;i1=n; }
                else if (s > v2) { v3=v2;i3=i2; v2=s;i2=n; }
                else             { v3=s;i3=n; }
            }
        }
        float rq = qrn[tBase + lane];
        v0 *= rq; v1 *= rq; v2 *= rq; v3 *= rq;       // cosine scores
        float e0 = 1.0f;                              // v0 is the max
        float e1 = __expf(v1 - v0);
        float e2 = __expf(v2 - v0);
        float e3 = __expf(v3 - v0);
        float rs = 1.0f / (e0 + e1 + e2 + e3);
        selIdx[wave][lane][0] = i0; selW[wave][lane][0] = e0 * rs;
        selIdx[wave][lane][1] = i1; selW[wave][lane][1] = e1 * rs;
        selIdx[wave][lane][2] = i2; selW[wave][lane][2] = e2 * rs;
        selIdx[wave][lane][3] = i3; selW[wave][lane][3] = e3 * rs;
    }
    __syncthreads();

    for (int m = 0; m < 16; ++m) {        // weighted gather, lanes split H
        const int t = tBase + m;
        float4 av[16];
        #pragma unroll
        for (int c = 0; c < 16; ++c) av[c] = make_float4(0.f, 0.f, 0.f, 0.f);
        #pragma unroll
        for (int k = 0; k < 4; ++k) {
            const int   id = selIdx[wave][m][k];
            const float w  = selW[wave][m][k];
            const float4* cr = (const float4*)(contents + (size_t)id * H_DIM +
                                               lane * 64);
            #pragma unroll
            for (int c = 0; c < 16; ++c) {
                float4 x = cr[c];
                av[c].x = fmaf(w, x.x, av[c].x);
                av[c].y = fmaf(w, x.y, av[c].y);
                av[c].z = fmaf(w, x.z, av[c].z);
                av[c].w = fmaf(w, x.w, av[c].w);
            }
        }
        unsigned short* op = memb + (size_t)t * H_DIM + lane * 64;
        #pragma unroll
        for (int c = 0; c < 16; c += 2) {
            BF16x8 pk;
            pk.s[0] = f2b(av[c].x);   pk.s[1] = f2b(av[c].y);
            pk.s[2] = f2b(av[c].z);   pk.s[3] = f2b(av[c].w);
            pk.s[4] = f2b(av[c+1].x); pk.s[5] = f2b(av[c+1].y);
            pk.s[6] = f2b(av[c+1].z); pk.s[7] = f2b(av[c+1].w);
            *(uint4*)(op + c * 4) = pk.u;
        }
    }
}

// ---------------------------------------------------------------------------
extern "C" void kernel_launch(void* const* d_in, const int* in_sizes, int n_in,
                              void* d_out, int out_size, void* d_ws, size_t ws_size,
                              hipStream_t stream) {
    (void)in_sizes; (void)n_in; (void)out_size; (void)ws_size;
    const float* x         = (const float*)d_in[0];   // [4,4096,2048]
    const float* addresses = (const float*)d_in[1];   // [128,2048]
    const float* contents  = (const float*)d_in[2];   // [128,2048]
    const float* W_addr    = (const float*)d_in[3];   // [2048,2048]
    const float* W_read    = (const float*)d_in[4];   // [2048,2048]
    float* out = (float*)d_out;                       // [4,4096,2048]

    char* ws = (char*)d_ws;
    unsigned short* q_bf     = (unsigned short*)(ws + 0);
    unsigned short* mem_bf   = (unsigned short*)(ws + 67108864);
    unsigned short* x_bf     = (unsigned short*)(ws + 134217728);
    unsigned short* waddr_bf = (unsigned short*)(ws + 201326592);
    unsigned short* wread_bf = (unsigned short*)(ws + 209715200);
    unsigned short* knorm_bf = (unsigned short*)(ws + 218103808);
    float*          qrn      = (float*)        (ws + 218628096);

    // operands -> bf16 (row-major layouts are WMMA-fragment friendly)
    cvt_f32_bf16_kernel<<<(H_DIM * H_DIM) / (256 * 8), 256, 0, stream>>>(W_addr, waddr_bf);
    cvt_f32_bf16_kernel<<<(H_DIM * H_DIM) / (256 * 8), 256, 0, stream>>>(W_read, wread_bf);
    cvt_f32_bf16_kernel<<<(T_TOK * H_DIM) / (256 * 8), 256, 0, stream>>>(x, x_bf);

    // normalize addresses -> bf16
    addr_norm_kernel<<<N_MEM, 256, 0, stream>>>(addresses, knorm_bf);

    // q = x @ W_addr^T  (bf16 WMMA, bf16 output)
    gemm_wmma_kernel<true><<<dim3(H_DIM / 128, T_TOK / 128), 256, 0, stream>>>(
        x_bf, waddr_bf, (void*)q_bf);

    // per-token 1/||q||
    q_rnorm_kernel<<<T_TOK / 8, 256, 0, stream>>>(q_bf, qrn);

    // scores + top-4 + softmax + gather -> mem (bf16)
    attend_kernel<<<T_TOK / 128, 256, 0, stream>>>(q_bf, knorm_bf, qrn, contents, mem_bf);

    // out = mem @ W_read^T  (bf16 WMMA, f32 output)
    gemm_wmma_kernel<false><<<dim3(H_DIM / 128, T_TOK / 128), 256, 0, stream>>>(
        mem_bf, wread_bf, out);
}
